// RouterKT_10574209483087
// MI455X (gfx1250) — matrix operand order repaired and checked by reading, not compiled, not run on hardware
//
#include <hip/hip_runtime.h>

typedef unsigned short u16;
typedef __attribute__((ext_vector_type(16))) __bf16 v16bf;
typedef __attribute__((ext_vector_type(8)))  __bf16 v8bf;
typedef __attribute__((ext_vector_type(8)))  float  v8f;

union FragBF { v16bf v; v8bf h[2]; };

// ---------------------------------------------------------------- constants
#define BATCH 16
#define SEQ   512
#define DIM   512
#define HEADS 8
#define DK    64
#define DFF   2048
#define LAYERS 6
#define MROWS (BATCH*SEQ)     // 8192
#define NROUTE 6              // H - HS
#define SPAD  520             // padded score row stride (floats / u16s)

// ---------------------------------------------------------------- helpers
__device__ __forceinline__ u16 f2bf(float f) {
    unsigned u = __float_as_uint(f);
    unsigned r = u + 0x7FFFu + ((u >> 16) & 1u);   // round-to-nearest-even
    return (u16)(r >> 16);
}

// A fragment: 16x32 bf16, row-major source, lane m = L&15, halves split K by 8
__device__ __forceinline__ v16bf ldA(const u16* base, int ld) {
    int lane = threadIdx.x & 31;
    int m  = lane & 15;
    int ha = lane >> 4;
    FragBF f;
    f.h[0] = *reinterpret_cast<const v8bf*>(base + (size_t)m * ld + ha * 8);
    f.h[1] = *reinterpret_cast<const v8bf*>(base + (size_t)m * ld + ha * 8 + 16);
    return f.v;
}
// B fragment: 32x16 bf16 from transposed ([n][k]) source, lane n = L&15,
// halves split K by 8 (same k order as A fragment)
__device__ __forceinline__ v16bf ldB(const u16* baseT, int ld) {
    int lane = threadIdx.x & 31;
    int n  = lane & 15;
    int hh = lane >> 4;
    FragBF f;
    f.h[0] = *reinterpret_cast<const v8bf*>(baseT + (size_t)n * ld + hh * 8);
    f.h[1] = *reinterpret_cast<const v8bf*>(baseT + (size_t)n * ld + hh * 8 + 16);
    return f.v;
}
#define WMMA_BF16(a, b, c) \
    __builtin_amdgcn_wmma_f32_16x16x32_bf16(false, (a), false, (b), (short)0, (c), false, false)

// ---------------------------------------------------------------- misc kernels
__global__ void init_act_kernel(const float* __restrict__ src,
                                float* __restrict__ dstF,
                                u16* __restrict__ dstB, int n) {
    int i = blockIdx.x * blockDim.x + threadIdx.x;
    if (i < n) { float v = src[i]; dstF[i] = v; dstB[i] = f2bf(v); }
}

__global__ void zero_kernel(float* p, int n) {
    int i = blockIdx.x * blockDim.x + threadIdx.x;
    if (i < n) p[i] = 0.0f;
}

// Transpose [K,N] fp32 -> [N,K] bf16 (blockIdx.z batches with stride K*N)
__global__ __launch_bounds__(256) void wtrans_kernel(const float* __restrict__ W,
                                                     u16* __restrict__ WT,
                                                     int K, int N) {
    __shared__ float t[32][33];
    size_t zoff = (size_t)blockIdx.z * K * N;
    W  += zoff;
    WT += zoff;
    int kb = blockIdx.x * 32, nb = blockIdx.y * 32;
    int tx = threadIdx.x & 31, ty = threadIdx.x >> 5;   // 32 x 8
#pragma unroll
    for (int i = 0; i < 4; i++)
        t[ty + i * 8][tx] = W[(size_t)(kb + ty + i * 8) * N + nb + tx];
    __syncthreads();
#pragma unroll
    for (int i = 0; i < 4; i++)
        WT[(size_t)(nb + ty + i * 8) * K + kb + tx] = f2bf(t[tx][ty + i * 8]);
}

// ---------------------------------------------------------------- bf16 WMMA GEMM
// C[M,N] = A[M,K]@W[K,N] + bias  (W given transposed [N,K] bf16)
// block = 256 thr (8 waves); tile 128(M) x 64(N); wave owns 16 rows, 4 n-subtiles
// Software-pipelined with ping-pong fragment buffers (2x K unroll): loads for
// one buffer overlap the 4 chained WMMAs on the other, and the loop-carried
// fragments live in fixed registers (no rotation copies at the back-edge).
__global__ __launch_bounds__(256) void gemm_bf16_kernel(
        const u16* __restrict__ A, const u16* __restrict__ WT,
        const float* __restrict__ bias,
        float* __restrict__ Cf, u16* __restrict__ Cb,
        int M, int N, int K, int relu) {
    int wid  = threadIdx.x >> 5;
    int lane = threadIdx.x & 31;
    int m0 = blockIdx.x * 128 + wid * 16;
    int n0 = blockIdx.y * 64;
    int nn = lane & 15, hh = lane >> 4;

    v8f acc[4];
#pragma unroll
    for (int t = 0; t < 4; t++) {
        float bv = bias ? bias[n0 + t * 16 + nn] : 0.0f;
#pragma unroll
        for (int i = 0; i < 8; i++) acc[t][i] = bv;
    }
    const u16* Abase = A + (size_t)m0 * K;
    const u16* Bbase = WT + (size_t)n0 * K;

    // buffer P primed with step 0
    v16bf aP = ldA(Abase, K);
    v16bf bP[4];
#pragma unroll
    for (int t = 0; t < 4; t++) bP[t] = ldB(Bbase + (size_t)t * 16 * K, K);

    int nsteps = K >> 5;            // 16 (K=512) or 64 (K=2048), always even
    for (int s = 0; s < nsteps - 2; s += 2) {
        v16bf aQ = ldA(Abase + (s + 1) * 32, K);
        v16bf bQ[4];
#pragma unroll
        for (int t = 0; t < 4; t++) bQ[t] = ldB(Bbase + (size_t)t * 16 * K + (s + 1) * 32, K);
#pragma unroll
        for (int t = 0; t < 4; t++) acc[t] = WMMA_BF16(aP, bP[t], acc[t]);
        aP = ldA(Abase + (s + 2) * 32, K);
#pragma unroll
        for (int t = 0; t < 4; t++) bP[t] = ldB(Bbase + (size_t)t * 16 * K + (s + 2) * 32, K);
#pragma unroll
        for (int t = 0; t < 4; t++) acc[t] = WMMA_BF16(aQ, bQ[t], acc[t]);
    }
    {   // final two steps (peeled)
        v16bf aQ = ldA(Abase + (nsteps - 1) * 32, K);
        v16bf bQ[4];
#pragma unroll
        for (int t = 0; t < 4; t++) bQ[t] = ldB(Bbase + (size_t)t * 16 * K + (nsteps - 1) * 32, K);
#pragma unroll
        for (int t = 0; t < 4; t++) acc[t] = WMMA_BF16(aP, bP[t], acc[t]);
#pragma unroll
        for (int t = 0; t < 4; t++) acc[t] = WMMA_BF16(aQ, bQ[t], acc[t]);
    }

#pragma unroll
    for (int t = 0; t < 4; t++) {
#pragma unroll
        for (int i = 0; i < 8; i++) {
            float v = acc[t][i];
            if (relu) v = v > 0.0f ? v : 0.0f;
            size_t idx = (size_t)(m0 + i + 8 * hh) * N + n0 + t * 16 + nn;
            if (Cf) Cf[idx] = v;
            if (Cb) Cb[idx] = f2bf(v);
        }
    }
}

// ---------------------------------------------------------------- head routing
// gates = softmax(qlin @ wg) over 6 routed heads; keep top-2; atomic mean acc
__global__ void gates_kernel(const float* __restrict__ qlin,
                             const float* __restrict__ wg,
                             float* __restrict__ racc) {
    int idx = blockIdx.x * blockDim.x + threadIdx.x;   // b*SEQ + s
    if (idx >= MROWS) return;
    const float* qr = qlin + (size_t)idx * DIM;
    float g[NROUTE];
#pragma unroll
    for (int j = 0; j < NROUTE; j++) g[j] = 0.0f;
    for (int d = 0; d < DIM; d++) {
        float x = qr[d];
        const float* w = wg + d * NROUTE;
#pragma unroll
        for (int j = 0; j < NROUTE; j++) g[j] += x * w[j];
    }
    float mx = g[0];
#pragma unroll
    for (int j = 1; j < NROUTE; j++) mx = fmaxf(mx, g[j]);
    float s = 0.0f;
#pragma unroll
    for (int j = 0; j < NROUTE; j++) { g[j] = __expf(g[j] - mx); s += g[j]; }
    float inv = 1.0f / s;
#pragma unroll
    for (int j = 0; j < NROUTE; j++) g[j] *= inv;
    int i1 = 0;
#pragma unroll
    for (int j = 1; j < NROUTE; j++) if (g[j] > g[i1]) i1 = j;   // first max
    int i2 = (i1 == 0) ? 1 : 0;
#pragma unroll
    for (int j = 0; j < NROUTE; j++) if (j != i1 && g[j] > g[i2]) i2 = j;
    int b = idx >> 9;   // / SEQ
    atomicAdd(&racc[b * NROUTE + i1], g[i1]);
    atomicAdd(&racc[b * NROUTE + i2], g[i2]);
}

__global__ void rmask_fin_kernel(const float* __restrict__ racc,
                                 float* __restrict__ rmask) {
    int i = threadIdx.x;
    if (i < BATCH * HEADS) {
        int b = i >> 3, h = i & 7;
        rmask[i] = (h < 2) ? 1.0f : racc[b * NROUTE + (h - 2)] * (1.0f / (float)SEQ);
    }
}

// ---------------------------------------------------------------- fused attention
// one block per (b, h, 16-row q tile); 128 thr (4 waves)
// scores = (Q Q^T)/8 with causal mask (diag=0 or -1), row q==0 forced uniform,
// softmax, ctx = attn @ V, scaled by rmask[b,h], written bf16
__global__ __launch_bounds__(128) void attn_kernel(
        const u16* __restrict__ qb,     // qlin bf16  [B*S, D]
        const u16* __restrict__ vT,     // V^T bf16   [B*D, S]
        const float* __restrict__ rmask,
        u16* __restrict__ ctx,          // bf16 [B*S, D]
        int diag) {
    __shared__ float sc[16][SPAD];
    __shared__ u16   at[16][SPAD];
    int wid  = threadIdx.x >> 5;
    int lane = threadIdx.x & 31;
    int qt = blockIdx.x & 31;            // S/16 = 32 tiles
    int h  = (blockIdx.x >> 5) & 7;
    int b  = blockIdx.x >> 8;
    int q0 = qt * 16;

    const u16* Qbase = qb + ((size_t)(b * SEQ + q0) * DIM) + h * DK;
    v16bf a0 = ldA(Qbase,      DIM);
    v16bf a1 = ldA(Qbase + 32, DIM);

    // ---- phase 1: scores (causal tiles only), waves stride over k-tiles
    for (int kt = wid; kt <= qt; kt += 4) {
        const u16* Kb = qb + ((size_t)(b * SEQ + kt * 16) * DIM) + h * DK;
        v8f acc;
#pragma unroll
        for (int i = 0; i < 8; i++) acc[i] = 0.0f;
        v16bf b0 = ldB(Kb,      DIM);
        v16bf b1 = ldB(Kb + 32, DIM);
        acc = WMMA_BF16(a0, b0, acc);
        acc = WMMA_BF16(a1, b1, acc);
        int nn = lane & 15, hh = lane >> 4;
#pragma unroll
        for (int i = 0; i < 8; i++)
            sc[i + 8 * hh][kt * 16 + nn] = acc[i] * 0.125f;   // 1/sqrt(64)
    }
    __syncthreads();

    // ---- phase 2: masked softmax, 4 rows per wave
    u16 uni = f2bf(1.0f / (float)SEQ);
    for (int rr = 0; rr < 4; rr++) {
        int r = wid * 4 + rr;
        int q = q0 + r;
        if (q == 0) {   // pad_zero row: scores all 0 -> uniform attention
            for (int c = lane; c < SEQ; c += 32) at[r][c] = uni;
            continue;
        }
        int maxc = q + diag;   // diag = 0 (j<=i) or -1 (j<i)
        float mx = -1e30f;
        for (int c = lane; c < SEQ; c += 32)
            if (c <= maxc) mx = fmaxf(mx, sc[r][c]);
#pragma unroll
        for (int o = 16; o; o >>= 1) mx = fmaxf(mx, __shfl_xor(mx, o, 32));
        float sum = 0.0f;
        for (int c = lane; c < SEQ; c += 32) {
            float e = (c <= maxc) ? __expf(sc[r][c] - mx) : 0.0f;
            sc[r][c] = e;
            sum += e;
        }
#pragma unroll
        for (int o = 16; o; o >>= 1) sum += __shfl_xor(sum, o, 32);
        float inv = 1.0f / sum;
        for (int c = lane; c < SEQ; c += 32) at[r][c] = f2bf(sc[r][c] * inv);
    }
    __syncthreads();

    // ---- phase 3: ctx = attn @ V ; wave wid owns dk n-tile wid (16 cols)
    // Pipelined: next vT fragment fetched while current WMMA runs
    int ksend = (qt == 0) ? SEQ : (((q0 + 16) + 31) & ~31);   // row0 needs full range
    v8f acc;
#pragma unroll
    for (int i = 0; i < 8; i++) acc[i] = 0.0f;
    const u16* Vb = vT + (size_t)(b * DIM + h * DK + wid * 16) * SEQ;
    v16bf bcur = ldB(Vb, SEQ);
    for (int ks = 0; ks < ksend; ks += 32) {
        v16bf a = ldA(&at[0][ks], SPAD);
        v16bf bn = bcur;
        if (ks + 32 < ksend) bn = ldB(Vb + ks + 32, SEQ);
        acc = WMMA_BF16(a, bcur, acc);
        bcur = bn;
    }
    float rm = rmask[b * HEADS + h];
    int nn = lane & 15, hh = lane >> 4;
#pragma unroll
    for (int i = 0; i < 8; i++)
        ctx[(size_t)(b * SEQ + q0 + i + 8 * hh) * DIM + h * DK + wid * 16 + nn] =
            f2bf(acc[i] * rm);
}

// ---------------------------------------------------------------- residual + LN
// x = LN(x + add); one wave per row; writes fp32 stream and bf16 stream
__global__ __launch_bounds__(256) void ln_res_kernel(
        float* __restrict__ x, const float* __restrict__ add,
        const float* __restrict__ g, const float* __restrict__ bta,
        u16* __restrict__ xb) {
    int wid  = threadIdx.x >> 5;
    int lane = threadIdx.x & 31;
    int row = blockIdx.x * 8 + wid;
    float* xr = x + (size_t)row * DIM;
    const float* ar = add + (size_t)row * DIM;
    float v[16];
    float s = 0.0f;
#pragma unroll
    for (int k = 0; k < 16; k++) {
        v[k] = xr[lane + 32 * k] + ar[lane + 32 * k];
        s += v[k];
    }
#pragma unroll
    for (int o = 16; o; o >>= 1) s += __shfl_xor(s, o, 32);
    float mu = s * (1.0f / (float)DIM);
    float vs = 0.0f;
#pragma unroll
    for (int k = 0; k < 16; k++) { float d = v[k] - mu; vs += d * d; }
#pragma unroll
    for (int o = 16; o; o >>= 1) vs += __shfl_xor(vs, o, 32);
    float inv = rsqrtf(vs * (1.0f / (float)DIM) + 1e-5f);
#pragma unroll
    for (int k = 0; k < 16; k++) {
        int d = lane + 32 * k;
        float o = g[d] * (v[k] - mu) * inv + bta[d];
        xr[d] = o;
        xb[(size_t)row * DIM + d] = f2bf(o);
    }
}

// ---------------------------------------------------------------- host
extern "C" void kernel_launch(void* const* d_in, const int* in_sizes, int n_in,
                              void* d_out, int out_size, void* d_ws, size_t ws_size,
                              hipStream_t stream) {
    (void)in_sizes; (void)n_in; (void)out_size; (void)ws_size;
    const float* question    = (const float*)d_in[0];
    const float* interaction = (const float*)d_in[1];
    const float* wq  = (const float*)d_in[2];
    const float* bq  = (const float*)d_in[3];
    const float* wv  = (const float*)d_in[4];
    const float* bv  = (const float*)d_in[5];
    const float* wo  = (const float*)d_in[6];
    const float* bo  = (const float*)d_in[7];
    const float* wg  = (const float*)d_in[8];
    const float* ln1g = (const float*)d_in[9];
    const float* ln1b = (const float*)d_in[10];
    const float* fw1 = (const float*)d_in[11];
    const float* fb1 = (const float*)d_in[12];
    const float* fw2 = (const float*)d_in[13];
    const float* fb2 = (const float*)d_in[14];
    const float* ln2g = (const float*)d_in[15];
    const float* ln2b = (const float*)d_in[16];

    // ---- workspace layout
    char* ws = (char*)d_ws;
    size_t off = 0;
    auto alloc = [&](size_t bytes) -> char* {
        char* p = ws + off;
        off += (bytes + 255) & ~(size_t)255;
        return p;
    };
    const size_t MD = (size_t)MROWS * DIM;
    u16* wqT  = (u16*)alloc((size_t)LAYERS * DIM * DIM * 2);
    u16* wvT  = (u16*)alloc((size_t)LAYERS * DIM * DIM * 2);
    u16* woT  = (u16*)alloc((size_t)LAYERS * DIM * DIM * 2);
    u16* fw1T = (u16*)alloc((size_t)LAYERS * DFF * DIM * 2);
    u16* fw2T = (u16*)alloc((size_t)LAYERS * DIM * DFF * 2);
    float* x_f  = (float*)alloc(MD * 4);
    u16*   x_b  = (u16*)alloc(MD * 2);
    float* y_f  = (float*)alloc(MD * 4);
    u16*   y_b  = (u16*)alloc(MD * 2);
    float* q_f  = (float*)alloc(MD * 4);
    u16*   q_b  = (u16*)alloc(MD * 2);
    float* v_f  = (float*)alloc(MD * 4);
    u16*   vT   = (u16*)alloc(MD * 2);
    u16*   ctx_b = (u16*)alloc(MD * 2);
    float* o_f  = (float*)alloc(MD * 4);
    u16*   h_b  = (u16*)alloc((size_t)MROWS * DFF * 2);
    float* racc  = (float*)alloc(BATCH * NROUTE * 4);
    float* rmask = (float*)alloc(BATCH * HEADS * 4);

    // ---- weight transposes (fp32 -> bf16 [N,K]) for all layers at once
    wtrans_kernel<<<dim3(DIM / 32, DIM / 32, LAYERS), 256, 0, stream>>>(wq, wqT, DIM, DIM);
    wtrans_kernel<<<dim3(DIM / 32, DIM / 32, LAYERS), 256, 0, stream>>>(wv, wvT, DIM, DIM);
    wtrans_kernel<<<dim3(DIM / 32, DIM / 32, LAYERS), 256, 0, stream>>>(wo, woT, DIM, DIM);
    wtrans_kernel<<<dim3(DIM / 32, DFF / 32, LAYERS), 256, 0, stream>>>(fw1, fw1T, DIM, DFF);
    wtrans_kernel<<<dim3(DFF / 32, DIM / 32, LAYERS), 256, 0, stream>>>(fw2, fw2T, DFF, DIM);

    // ---- initial activation streams
    init_act_kernel<<<(int)(MD / 256), 256, 0, stream>>>(question, x_f, x_b, (int)MD);
    init_act_kernel<<<(int)(MD / 256), 256, 0, stream>>>(interaction, y_f, y_b, (int)MD);

    auto run_block = [&](int l, float* sF, u16* sB, const u16* xvB, int diag, bool ffn) {
        // q = s @ wq + bq  (fp32 for routing, bf16 for attention)
        gemm_bf16_kernel<<<dim3(MROWS / 128, DIM / 64), 256, 0, stream>>>(
            sB, wqT + (size_t)l * DIM * DIM, bq + (size_t)l * DIM,
            q_f, q_b, MROWS, DIM, DIM, 0);
        // v = xv @ wv + bv
        gemm_bf16_kernel<<<dim3(MROWS / 128, DIM / 64), 256, 0, stream>>>(
            xvB, wvT + (size_t)l * DIM * DIM, bv + (size_t)l * DIM,
            v_f, nullptr, MROWS, DIM, DIM, 0);
        // V^T per batch: [S,D] -> [D,S] bf16
        wtrans_kernel<<<dim3(SEQ / 32, DIM / 32, BATCH), 256, 0, stream>>>(v_f, vT, SEQ, DIM);
        // routing mask
        zero_kernel<<<1, BATCH * NROUTE, 0, stream>>>(racc, BATCH * NROUTE);
        gates_kernel<<<MROWS / 256, 256, 0, stream>>>(q_f, wg + (size_t)l * DIM * NROUTE, racc);
        rmask_fin_kernel<<<1, 128, 0, stream>>>(racc, rmask);
        // fused attention
        attn_kernel<<<BATCH * HEADS * (SEQ / 16), 128, 0, stream>>>(q_b, vT, rmask, ctx_b, diag);
        // output projection
        gemm_bf16_kernel<<<dim3(MROWS / 128, DIM / 64), 256, 0, stream>>>(
            ctx_b, woT + (size_t)l * DIM * DIM, bo + (size_t)l * DIM,
            o_f, nullptr, MROWS, DIM, DIM, 0);
        // s = LN(s + o)
        ln_res_kernel<<<MROWS / 8, 256, 0, stream>>>(
            sF, o_f, ln1g + (size_t)l * DIM, ln1b + (size_t)l * DIM, sB);
        if (ffn) {
            gemm_bf16_kernel<<<dim3(MROWS / 128, DFF / 64), 256, 0, stream>>>(
                sB, fw1T + (size_t)l * DFF * DIM, fb1 + (size_t)l * DFF,
                nullptr, h_b, MROWS, DFF, DIM, 1);
            gemm_bf16_kernel<<<dim3(MROWS / 128, DIM / 64), 256, 0, stream>>>(
                h_b, fw2T + (size_t)l * DIM * DFF, fb2 + (size_t)l * DIM,
                o_f, nullptr, MROWS, DIM, DFF, 0);
            ln_res_kernel<<<MROWS / 8, 256, 0, stream>>>(
                sF, o_f, ln2g + (size_t)l * DIM, ln2b + (size_t)l * DIM, sB);
        }
    };

    // knowledge encoder (stream = y)
    run_block(0, y_f, y_b, y_b, 0, false);
    run_block(1, y_f, y_b, y_b, 0, false);
    // question encoder (stream = x), alternating self / cross+FFN
    run_block(2, x_f, x_b, x_b, 0, false);
    run_block(3, x_f, x_b, y_b, -1, true);
    run_block(4, x_f, x_b, x_b, 0, false);
    run_block(5, x_f, x_b, y_b, -1, true);

    hipMemcpyAsync(d_out, x_f, MD * sizeof(float), hipMemcpyDeviceToDevice, stream);
}
